// BahdanauAttention_54065048322491
// MI455X (gfx1250) — matrix operand order repaired and checked by reading, not compiled, not run on hardware
//
#include <hip/hip_runtime.h>
#include <hip/hip_bf16.h>
#include <math.h>

#define NB 32
#define NS 2048
#define NH 1024
#define NCH 8          // context S-chunks

typedef __attribute__((ext_vector_type(16))) __bf16 v16bf;
typedef __attribute__((ext_vector_type(8)))  __bf16 v8bf;
typedef __attribute__((ext_vector_type(8)))  float  v8f;

#define HCHUNK 32
#define LROW 40   // 32 bf16 + 8 pad -> 80B row stride (16B aligned, bank-spread)

__device__ __forceinline__ v8bf cvt8(float4 x, float4 y) {
  v8bf r;
  r[0] = (__bf16)x.x; r[1] = (__bf16)x.y; r[2] = (__bf16)x.z; r[3] = (__bf16)x.w;
  r[4] = (__bf16)y.x; r[5] = (__bf16)y.y; r[6] = (__bf16)y.z; r[7] = (__bf16)y.w;
  return r;
}

// ---------- Kernel 1: qb[b][o] = query[b,:]·Wa_w[o,:] + Wa_b[o] + Ua_b[o] ----------
__global__ __launch_bounds__(128) void bah_qproj(
    const float* __restrict__ query, const float* __restrict__ Wa_w,
    const float* __restrict__ Wa_b,  const float* __restrict__ Ua_b,
    float* __restrict__ qb)
{
  const int b = blockIdx.y;
  const int o = blockIdx.x * 128 + threadIdx.x;
  const float4* q = (const float4*)(query + (size_t)b * NH);
  const float4* w = (const float4*)(Wa_w  + (size_t)o * NH);
  float s = 0.f;
#pragma unroll 4
  for (int i = 0; i < NH / 4; ++i) {
    float4 a = q[i], c = w[i];
    s += a.x * c.x + a.y * c.y + a.z * c.z + a.w * c.w;
  }
  qb[b * NH + o] = s + Wa_b[o] + Ua_b[o];
}

// ---------- Kernel 2: fused bf16-WMMA GEMM + tanh + Va dot -> scores[b][s] ----------
// block = 256 threads (8 wave32), covers 128 s-rows of one batch b.
// Double-buffered LDS staging (one barrier per k-step); branch-free tanh epilogue.
// Stored scores are shifted by the constant Sum_o(Va_w[o]) -> softmax-invariant.
__global__ __launch_bounds__(256, 2) void bah_scores(
    const float* __restrict__ keys, const float* __restrict__ Ua_w,
    const float* __restrict__ Va_w, const float* __restrict__ qb,
    float* __restrict__ scores)
{
  __shared__ __bf16 ldsA[2][128 * LROW];  // keys tile  [s=128][h=32]
  __shared__ __bf16 ldsB[2][128 * LROW];  // Ua_w tile  [o=128][h=32]

  const int tid  = threadIdx.x;
  const int b    = blockIdx.y;
  const int s0   = blockIdx.x * 128;
  const int lane = tid & 31;
  const int wv   = tid >> 5;       // wave id: owns s-rows [s0+wv*16, +16)
  const int mn   = lane & 15;      // M (A) / N (B,C) index
  const int kh   = lane >> 4;      // lane half -> K-half / M-half

  // staging assignment: thread -> (row, 16-col half)
  const int srow = tid >> 1;
  const int scol = (tid & 1) * 16;
  const int stoff = srow * LROW + scol;
  const size_t kbase = ((size_t)b * NS + s0 + srow) * NH + scol;
  const size_t ubase = (size_t)srow * NH + scol;

  const float* qbp = qb + (size_t)b * NH;

  float part[8];
#pragma unroll
  for (int r = 0; r < 8; ++r) part[r] = 0.f;

  // prologue: prefetch chunk (ob=0, h0=0) and stage into buffer 0
  float4 ka0, ka1, ka2, ka3, ua0, ua1, ua2, ua3;
  {
    const float4* gK = (const float4*)(keys + kbase);
    const float4* gU = (const float4*)(Ua_w + ubase);
    ka0 = gK[0]; ka1 = gK[1]; ka2 = gK[2]; ka3 = gK[3];
    ua0 = gU[0]; ua1 = gU[1]; ua2 = gU[2]; ua3 = gU[3];
  }
  *(v8bf*)&ldsA[0][stoff]     = cvt8(ka0, ka1);
  *(v8bf*)&ldsA[0][stoff + 8] = cvt8(ka2, ka3);
  *(v8bf*)&ldsB[0][stoff]     = cvt8(ua0, ua1);
  *(v8bf*)&ldsB[0][stoff + 8] = cvt8(ua2, ua3);
  int pcur = 0;

  for (int ob = 0; ob < NH / 128; ++ob) {          // 8 o-blocks of 128
    v8f acc[8];
#pragma unroll
    for (int ot = 0; ot < 8; ++ot) acc[ot] = 0.f;

    for (int hi = 0; hi < NH / HCHUNK; ++hi) {     // 32 k-steps
      const bool last = (ob == NH / 128 - 1) && (hi == NH / HCHUNK - 1);
      __syncthreads();   // buf[pcur] ready; previous reads of buf[pcur^1] done

      // prefetch next chunk while this one computes (uniform branch)
      if (!last) {
        int nob = ob, nh = (hi + 1) * HCHUNK;
        if (hi == NH / HCHUNK - 1) { nob = ob + 1; nh = 0; }
        const float4* gK = (const float4*)(keys + kbase + nh);
        const float4* gU = (const float4*)(Ua_w + (size_t)nob * 128 * NH + ubase + nh);
        ka0 = gK[0]; ka1 = gK[1]; ka2 = gK[2]; ka3 = gK[3];
        ua0 = gU[0]; ua1 = gU[1]; ua2 = gU[2]; ua3 = gU[3];
      }

      // A fragment: lane mn = row M, K-halves interleaved per ISA A-16bit table
      const __bf16* pa = &ldsA[pcur][(wv * 16 + mn) * LROW];
      v8bf alo = *(const v8bf*)(pa + 8 * kh);
      v8bf ahi = *(const v8bf*)(pa + 16 + 8 * kh);
      v16bf afrag;
#pragma unroll
      for (int i = 0; i < 8; ++i) { afrag[i] = alo[i]; afrag[i + 8] = ahi[i]; }

#pragma unroll
      for (int ot = 0; ot < 8; ++ot) {
        const __bf16* pb = &ldsB[pcur][(ot * 16 + mn) * LROW + 8 * kh];
        v8bf blo = *(const v8bf*)(pb);
        v8bf bhi = *(const v8bf*)(pb + 16);
        v16bf bfrag;
#pragma unroll
        for (int i = 0; i < 8; ++i) { bfrag[i] = blo[i]; bfrag[i + 8] = bhi[i]; }
        acc[ot] = __builtin_amdgcn_wmma_f32_16x16x32_bf16(
            false, afrag, false, bfrag, (short)0, acc[ot], false, false);
      }

      // stage next chunk into the other buffer
      if (!last) {
        __bf16* dA = &ldsA[pcur ^ 1][stoff];
        __bf16* dB = &ldsB[pcur ^ 1][stoff];
        *(v8bf*)(dA)     = cvt8(ka0, ka1);
        *(v8bf*)(dA + 8) = cvt8(ka2, ka3);
        *(v8bf*)(dB)     = cvt8(ua0, ua1);
        *(v8bf*)(dB + 8) = cvt8(ua2, ua3);
      }
      pcur ^= 1;
    }

    // epilogue: tanh(kproj+qb)·Va_w, branch-free:
    //   vw*tanh(x) = vw - 2*vw*rcp(exp(2x)+1); constant Sum(vw) dropped (softmax-invariant)
#pragma unroll
    for (int ot = 0; ot < 8; ++ot) {
      const int o = ob * 128 + ot * 16 + mn;
      const float qv2 = 2.f * qbp[o];
      const float vw2 = 2.f * Va_w[o];
#pragma unroll
      for (int r = 0; r < 8; ++r) {
        float u = __builtin_amdgcn_rcpf(__expf(fmaf(2.f, acc[ot][r], qv2)) + 1.f);
        part[r] = fmaf(-vw2, u, part[r]);
      }
    }
  }

  // reduce over N (16 lanes within each half), then store scores (deterministic)
#pragma unroll
  for (int r = 0; r < 8; ++r) {
    part[r] += __shfl_xor(part[r], 1, 32);
    part[r] += __shfl_xor(part[r], 2, 32);
    part[r] += __shfl_xor(part[r], 4, 32);
    part[r] += __shfl_xor(part[r], 8, 32);
  }
  if (mn == 0) {
    const int sbase = s0 + wv * 16 + kh * 8;   // C layout: row M = r + 8*kh
#pragma unroll
    for (int r = 0; r < 8; ++r)
      scores[(size_t)b * NS + sbase + r] = part[r];
  }
}

// ---------- Kernel 3a: softmax over S -> weights ----------
__global__ __launch_bounds__(256) void bah_softmax(
    const float* __restrict__ scores, float* __restrict__ out_w)
{
  __shared__ float sw[NS];
  __shared__ float red[8];
  const int b = blockIdx.x, tid = threadIdx.x, lane = tid & 31, wv = tid >> 5;
  const float* sc = scores + (size_t)b * NS;

  float lmax = -3.402823466e38f;
  for (int s = tid; s < NS; s += 256) { float v = sc[s]; sw[s] = v; lmax = fmaxf(lmax, v); }
#pragma unroll
  for (int off = 16; off; off >>= 1) lmax = fmaxf(lmax, __shfl_xor(lmax, off, 32));
  if (lane == 0) red[wv] = lmax;
  __syncthreads();
  float bmax = red[0];
#pragma unroll
  for (int i = 1; i < 8; ++i) bmax = fmaxf(bmax, red[i]);
  __syncthreads();

  float lsum = 0.f;
  for (int s = tid; s < NS; s += 256) { float e = __expf(sw[s] - bmax); sw[s] = e; lsum += e; }
#pragma unroll
  for (int off = 16; off; off >>= 1) lsum += __shfl_xor(lsum, off, 32);
  if (lane == 0) red[wv] = lsum;
  __syncthreads();
  float tot = 0.f;
#pragma unroll
  for (int i = 0; i < 8; ++i) tot += red[i];
  const float inv = 1.f / tot;

  for (int s = tid; s < NS; s += 256) out_w[(size_t)b * NS + s] = sw[s] * inv;
}

// ---------- Kernel 3b: partial context over an S-chunk (256 s-rows) ----------
__global__ __launch_bounds__(256) void bah_ctx_partial(
    const float* __restrict__ keys, const float* __restrict__ w,
    float* __restrict__ partial)
{
  const int c = blockIdx.x, b = blockIdx.y, tid = threadIdx.x;
  const int SC = NS / NCH;   // 256
  __shared__ float sw[NS / NCH];
  sw[tid] = w[(size_t)b * NS + c * SC + tid];
  __syncthreads();

  float4 acc = {0.f, 0.f, 0.f, 0.f};
  const float4* kp = (const float4*)(keys + ((size_t)b * NS + (size_t)c * SC) * NH) + tid;
  for (int s = 0; s < SC; ++s) {
    const float wv = sw[s];
    float4 k = kp[(size_t)s * (NH / 4)];
    acc.x += wv * k.x; acc.y += wv * k.y; acc.z += wv * k.z; acc.w += wv * k.w;
  }
  ((float4*)(partial + (size_t)(b * NCH + c) * NH))[tid] = acc;
}

// ---------- Kernel 3c: reduce partial contexts ----------
__global__ __launch_bounds__(256) void bah_ctx_reduce(
    const float* __restrict__ partial, float* __restrict__ out_ctx)
{
  const int b = blockIdx.x, tid = threadIdx.x;
  float4 acc = {0.f, 0.f, 0.f, 0.f};
  const float4* pp = (const float4*)(partial + (size_t)b * NCH * NH) + tid;
#pragma unroll
  for (int c = 0; c < NCH; ++c) {
    float4 p = pp[(size_t)c * (NH / 4)];
    acc.x += p.x; acc.y += p.y; acc.z += p.z; acc.w += p.w;
  }
  ((float4*)(out_ctx + (size_t)b * NH))[tid] = acc;
}

extern "C" void kernel_launch(void* const* d_in, const int* in_sizes, int n_in,
                              void* d_out, int out_size, void* d_ws, size_t ws_size,
                              hipStream_t stream) {
  const float* query = (const float*)d_in[0];
  const float* keys  = (const float*)d_in[1];
  const float* Wa_w  = (const float*)d_in[2];
  const float* Wa_b  = (const float*)d_in[3];
  const float* Ua_w  = (const float*)d_in[4];
  const float* Ua_b  = (const float*)d_in[5];
  const float* Va_w  = (const float*)d_in[6];
  // d_in[7] = Va_b: uniform shift on scores -> softmax-invariant, dropped.

  float* out_ctx = (float*)d_out;            // [32,1,1024]
  float* out_w   = (float*)d_out + NB * NH;  // [32,1,2048]
  float* qb      = (float*)d_ws;                       // [32,1024]
  float* scores  = (float*)d_ws + NB * NH;             // [32,2048]
  float* partial = (float*)d_ws + NB * NH + NB * NS;   // [32,NCH,1024]

  bah_qproj<<<dim3(NH / 128, NB), 128, 0, stream>>>(query, Wa_w, Wa_b, Ua_b, qb);
  bah_scores<<<dim3(NS / 128, NB), 256, 0, stream>>>(keys, Ua_w, Va_w, qb, scores);
  bah_softmax<<<NB, 256, 0, stream>>>(scores, out_w);
  bah_ctx_partial<<<dim3(NCH, NB), 256, 0, stream>>>(keys, out_w, partial);
  bah_ctx_reduce<<<NB, 256, 0, stream>>>(partial, out_ctx);
}
